// ComplexTransformerBlock_3539053051946
// MI455X (gfx1250) — matrix operand order repaired and checked by reading, not compiled, not run on hardware
//
#include <hip/hip_runtime.h>

// ---------------------------------------------------------------------------
// Problem constants (from the reference)
// ---------------------------------------------------------------------------
#define B_    4
#define S_    1024
#define L_    256
#define F_    512
#define CD_   768
#define NH_   8
#define DH_   64        // F / NH
#define E_    8
#define HID_  2048
#define NTOK  (B_ * S_)     // 4096
#define NCTX  (B_ * L_)     // 1024

typedef unsigned short bf_t;  // raw bf16 storage
typedef __attribute__((ext_vector_type(16))) __bf16 v16bf;
typedef __attribute__((ext_vector_type(8)))  float  v8f;
typedef int gcc_int4 __attribute__((vector_size(16)));   // matches builtin param

#define AS1 __attribute__((address_space(1)))
#define AS3 __attribute__((address_space(3)))

#if defined(__HIP_DEVICE_COMPILE__) && \
    __has_builtin(__builtin_amdgcn_global_load_async_to_lds_b128) && \
    __has_builtin(__builtin_amdgcn_s_wait_asynccnt)
#define USE_ASYNC_LDS 1
#endif

union Frag { uint4 q[2]; v16bf v; };

__device__ __forceinline__ bf_t f2bf(float f) {
    union { float f; unsigned u; } a; a.f = f;
    unsigned r = a.u + 0x7FFFu + ((a.u >> 16) & 1u);   // round-to-nearest-even
    return (bf_t)(r >> 16);
}

__device__ __forceinline__ v8f vzero8() {
    v8f z;
#pragma unroll
    for (int i = 0; i < 8; ++i) z[i] = 0.0f;
    return z;
}

// A fragment: 16x32 bf16, row-major [M][K] source.
// ISA layout: lanes 0-15 row m, VGPR0-3 K=0..7, VGPR4-7 K=16..23;
//             lanes 16-31 row m, VGPR0-3 K=8..15, VGPR4-7 K=24..31.
__device__ __forceinline__ v16bf frag_a(const bf_t* A, int lda, int m0, int k0, int lane) {
    int m = m0 + (lane & 15);
    int h = lane >> 4;
    const bf_t* p = A + (size_t)m * lda + k0 + h * 8;
    Frag f;
    f.q[0] = *(const uint4*)(p);        // K = h*8 .. h*8+7
    f.q[1] = *(const uint4*)(p + 16);   // K = 16 + h*8 .. 16 + h*8+7
    return f.v;
}

// B fragment: 32x16 bf16 from transposed storage Bt[N][K].
// ISA layout: lanes 0-15 col n, VGPR0-3 K=0..7, VGPR4-7 K=16..23;
//             lanes 16-31 col n, VGPR0-3 K=8..15, VGPR4-7 K=24..31.
__device__ __forceinline__ v16bf frag_b(const bf_t* Bt, int ldb, int n0, int k0, int lane) {
    int n = n0 + (lane & 15);
    int h = lane >> 4;
    const bf_t* p = Bt + (size_t)n * ldb + k0 + h * 8;
    Frag f;
    f.q[0] = *(const uint4*)(p);
    f.q[1] = *(const uint4*)(p + 16);
    return f.v;
}

// Negate all 16 packed bf16 values (sign-bit flip) -> gives A*(-B).
__device__ __forceinline__ v16bf fneg(v16bf x) {
    union { v16bf v; unsigned u[8]; } a; a.v = x;
#pragma unroll
    for (int i = 0; i < 8; ++i) a.u[i] ^= 0x80008000u;
    return a.v;
}

__device__ __forceinline__ v8f wmma_bf(v16bf a, v16bf b, v8f c) {
    return __builtin_amdgcn_wmma_f32_16x16x32_bf16(false, a, false, b, (short)0, c, false, false);
}

// ---------------------------------------------------------------------------
// Complex GEMM: (Cr + iCi) = (Ar + iAi) @ (Br + iBi) + bias, B pre-transposed.
// Register-blocked: one wave = 32x32 complex output (2x2 WMMA subtiles,
// 16 WMMAs per 32-K step, A/B fragments reused across subtiles).
// Block = 8 waves = 64 M x 128 N.  All shapes used divide exactly.
// ---------------------------------------------------------------------------
enum { CG_BIAS = 1, CG_ACC = 2, CG_MASK = 4, CG_STF32 = 8, CG_STBF = 16 };

__global__ __launch_bounds__(256) void cgemm_kernel(
    const bf_t* __restrict__ Ar, const bf_t* __restrict__ Ai, int lda,
    const bf_t* __restrict__ Btr, const bf_t* __restrict__ Bti, int ldb,
    const float* __restrict__ br, const float* __restrict__ bi,
    float* Cr, float* Ci, int ldc,
    bf_t* Crb, bf_t* Cib,
    const int* __restrict__ ridx, int expert,
    int M, int N, int K, int flags)
{
    int lane = threadIdx.x & 31;
    int wave = threadIdx.x >> 5;
    int wx = wave & 3;          // n-subgroup
    int wy = wave >> 2;         // m-subgroup
    int n0 = blockIdx.x * 128 + wx * 32;
    int m0 = blockIdx.y * 64  + wy * 32;
    if (n0 >= N || m0 >= M) return;

    v8f accr[2][2], acci[2][2];
#pragma unroll
    for (int mi = 0; mi < 2; ++mi)
#pragma unroll
        for (int ni = 0; ni < 2; ++ni) { accr[mi][ni] = vzero8(); acci[mi][ni] = vzero8(); }

    for (int k0 = 0; k0 < K; k0 += 32) {
        if (k0 + 32 < K) {   // global_prefetch_b8 of next K-slab
            __builtin_prefetch(Ar  + (size_t)(m0      + (lane & 15)) * lda + k0 + 32, 0, 1);
            __builtin_prefetch(Ar  + (size_t)(m0 + 16 + (lane & 15)) * lda + k0 + 32, 0, 1);
            __builtin_prefetch(Btr + (size_t)(n0      + (lane & 15)) * ldb + k0 + 32, 0, 1);
            __builtin_prefetch(Btr + (size_t)(n0 + 16 + (lane & 15)) * ldb + k0 + 32, 0, 1);
        }
        v16bf a_r[2], a_i[2], b_r[2], b_i[2], b_n[2];
#pragma unroll
        for (int mi = 0; mi < 2; ++mi) {
            a_r[mi] = frag_a(Ar, lda, m0 + mi * 16, k0, lane);
            a_i[mi] = frag_a(Ai, lda, m0 + mi * 16, k0, lane);
        }
#pragma unroll
        for (int ni = 0; ni < 2; ++ni) {
            b_r[ni] = frag_b(Btr, ldb, n0 + ni * 16, k0, lane);
            b_i[ni] = frag_b(Bti, ldb, n0 + ni * 16, k0, lane);
            b_n[ni] = fneg(b_i[ni]);
        }
#pragma unroll
        for (int mi = 0; mi < 2; ++mi)
#pragma unroll
            for (int ni = 0; ni < 2; ++ni) {
                accr[mi][ni] = wmma_bf(a_r[mi], b_r[ni], accr[mi][ni]);
                accr[mi][ni] = wmma_bf(a_i[mi], b_n[ni], accr[mi][ni]);   // - Ai@Bi
                acci[mi][ni] = wmma_bf(a_r[mi], b_i[ni], acci[mi][ni]);
                acci[mi][ni] = wmma_bf(a_i[mi], b_r[ni], acci[mi][ni]);
            }
    }

#pragma unroll
    for (int mi = 0; mi < 2; ++mi) {
        int mb = m0 + mi * 16 + ((lane >> 4) << 3);
#pragma unroll
        for (int ni = 0; ni < 2; ++ni) {
            int n = n0 + ni * 16 + (lane & 15);
            float bvr = (flags & CG_BIAS) ? br[n] : 0.0f;
            float bvi = (flags & CG_BIAS) ? bi[n] : 0.0f;
#pragma unroll
            for (int r = 0; r < 8; ++r) {
                int m = mb + r;
                float vr = accr[mi][ni][r] + bvr;
                float vi = acci[mi][ni][r] + bvi;
                if ((flags & CG_MASK) && ridx[m] != expert) continue;
                size_t o = (size_t)m * ldc + n;
                if (flags & CG_ACC)        { Cr[o] += vr; Ci[o] += vi; }
                else if (flags & CG_STF32) { Cr[o] = vr;  Ci[o] = vi;  }
                if (flags & CG_STBF)       { Crb[o] = f2bf(vr); Cib[o] = f2bf(vi); }
            }
        }
    }
}

// ---------------------------------------------------------------------------
// Attention scores: S = (Qr Kr^T + Qi Ki^T) * scale, per (b, head).
// K's natural [s][d] layout is exactly the Bt[N][K] fragment layout.
// ---------------------------------------------------------------------------
__global__ __launch_bounds__(256) void scores_kernel(
    const bf_t* __restrict__ Qr, const bf_t* __restrict__ Qi,
    const bf_t* __restrict__ Kr, const bf_t* __restrict__ Ki,
    float* __restrict__ Sc, int Sq, int Sk, int ldq, float scale)
{
    int lane = threadIdx.x & 31;
    int wave = threadIdx.x >> 5;
    int n0 = (blockIdx.x * 8 + wave) * 16;
    int m0 = blockIdx.y * 16;
    int bh = blockIdx.z, b = bh / NH_, h = bh % NH_;
    if (n0 >= Sk) return;

    const bf_t* qr = Qr + (size_t)b * Sq * ldq + h * DH_;
    const bf_t* qi = Qi + (size_t)b * Sq * ldq + h * DH_;
    const bf_t* kr = Kr + (size_t)b * Sk * ldq + h * DH_;
    const bf_t* ki = Ki + (size_t)b * Sk * ldq + h * DH_;

    v8f acc = vzero8();
    for (int k0 = 0; k0 < DH_; k0 += 32) {
        acc = wmma_bf(frag_a(qr, ldq, m0, k0, lane), frag_b(kr, ldq, n0, k0, lane), acc);
        acc = wmma_bf(frag_a(qi, ldq, m0, k0, lane), frag_b(ki, ldq, n0, k0, lane), acc);
    }

    float* out = Sc + (size_t)bh * Sq * Sk;
    int n  = n0 + (lane & 15);
    int mb = m0 + ((lane >> 4) << 3);
#pragma unroll
    for (int r = 0; r < 8; ++r) out[(size_t)(mb + r) * Sk + n] = acc[r] * scale;
}

// ---------------------------------------------------------------------------
// Mix: O = P @ V for both real & imag V, V pre-transposed to [F][Sk] per batch.
// 4 waves cover the 64 head dims; writes bf16 head-slices of [B][S][F].
// ---------------------------------------------------------------------------
__global__ __launch_bounds__(128) void mix_kernel(
    const bf_t* __restrict__ P,
    const bf_t* __restrict__ Vtr, const bf_t* __restrict__ Vti,
    bf_t* __restrict__ Or, bf_t* __restrict__ Oi,
    int Sq, int Sk)
{
    int lane = threadIdx.x & 31;
    int wave = threadIdx.x >> 5;
    int n0 = wave * 16;                       // 0..63 inside head
    int m0 = blockIdx.y * 16;
    int bh = blockIdx.z, b = bh / NH_, h = bh % NH_;

    const bf_t* p  = P   + (size_t)bh * Sq * Sk;
    const bf_t* vr = Vtr + (size_t)b * F_ * Sk + (size_t)h * DH_ * Sk;
    const bf_t* vi = Vti + (size_t)b * F_ * Sk + (size_t)h * DH_ * Sk;

    v8f ar = vzero8();
    v8f ai = vzero8();
    for (int k0 = 0; k0 < Sk; k0 += 32) {
        v16bf pf = frag_a(p, Sk, m0, k0, lane);
        ar = wmma_bf(pf, frag_b(vr, Sk, n0, k0, lane), ar);
        ai = wmma_bf(pf, frag_b(vi, Sk, n0, k0, lane), ai);
    }

    int n  = n0 + (lane & 15);
    int mb = m0 + ((lane >> 4) << 3);
#pragma unroll
    for (int r = 0; r < 8; ++r) {
        size_t o = (size_t)(b * Sq + mb + r) * F_ + h * DH_ + n;
        Or[o] = f2bf(ar[r]);
        Oi[o] = f2bf(ai[r]);
    }
}

// ---------------------------------------------------------------------------
// Row softmax (f32 in -> bf16 probs), one block per row.
// ---------------------------------------------------------------------------
__global__ __launch_bounds__(256) void softmax_kernel(
    const float* __restrict__ Sc, bf_t* __restrict__ P, int Sk)
{
    __shared__ float red[256];
    int t = threadIdx.x;
    size_t row = blockIdx.x;
    const float* s = Sc + row * Sk;
    bf_t* p = P + row * Sk;

    float mx = -3.4e38f;
    for (int i = t; i < Sk; i += 256) mx = fmaxf(mx, s[i]);
    red[t] = mx; __syncthreads();
    for (int st = 128; st > 0; st >>= 1) { if (t < st) red[t] = fmaxf(red[t], red[t + st]); __syncthreads(); }
    mx = red[0]; __syncthreads();

    float sum = 0.0f;
    for (int i = t; i < Sk; i += 256) sum += __expf(s[i] - mx);
    red[t] = sum; __syncthreads();
    for (int st = 128; st > 0; st >>= 1) { if (t < st) red[t] += red[t + st]; __syncthreads(); }
    float inv = 1.0f / red[0];

    for (int i = t; i < Sk; i += 256) p[i] = f2bf(__expf(s[i] - mx) * inv);
}

// ---------------------------------------------------------------------------
// Complex "amplitude" LayerNorm, one block per token. Emits f32 + bf16.
// ---------------------------------------------------------------------------
__global__ __launch_bounds__(256) void cln_kernel(
    const float* __restrict__ xr, const float* __restrict__ xi,
    const float* __restrict__ g,
    float* __restrict__ hr, float* __restrict__ hi,
    bf_t* __restrict__ hrb, bf_t* __restrict__ hib)
{
    __shared__ float red[256];
    int t = threadIdx.x;
    size_t base = (size_t)blockIdx.x * F_;

    float s = 0.0f;
    for (int i = t; i < F_; i += 256) {
        float r = xr[base + i], im = xi[base + i];
        s += sqrtf(r * r + im * im + 1e-6f);
    }
    red[t] = s; __syncthreads();
    for (int st = 128; st > 0; st >>= 1) { if (t < st) red[t] += red[t + st]; __syncthreads(); }
    float mean = red[0] / (float)F_;

    for (int i = t; i < F_; i += 256) {
        float r = xr[base + i], im = xi[base + i];
        float a = sqrtf(r * r + im * im + 1e-6f);
        float sc = g[i] * (a / (mean + 1e-6f)) / (a + 1e-6f);
        float vr = r * sc, vi = im * sc;
        hr[base + i] = vr;  hi[base + i] = vi;
        hrb[base + i] = f2bf(vr);  hib[base + i] = f2bf(vi);
    }
}

// ---------------------------------------------------------------------------
// Phase router: token expert index from circular mean of per-feature phases.
// ---------------------------------------------------------------------------
__global__ __launch_bounds__(256) void route_kernel(
    const float* __restrict__ hr, const float* __restrict__ hi, int* __restrict__ ridx)
{
    __shared__ float rs[256], rc[256];
    int t = threadIdx.x;
    size_t base = (size_t)blockIdx.x * F_;
    float ss = 0.0f, sc = 0.0f;
    for (int i = t; i < F_; i += 256) {
        float ph = atan2f(hi[base + i], hr[base + i]);
        ss += __sinf(ph);
        sc += __cosf(ph);
    }
    rs[t] = ss; rc[t] = sc; __syncthreads();
    for (int st = 128; st > 0; st >>= 1) {
        if (t < st) { rs[t] += rs[t + st]; rc[t] += rc[t + st]; }
        __syncthreads();
    }
    if (t == 0) {
        const float PI = 3.14159265358979323846f;
        float tp = atan2f(rs[0], rc[0]);
        int e = (int)floorf((tp + PI) / (2.0f * PI) * (float)E_);
        ridx[blockIdx.x] = e < 0 ? 0 : (e > E_ - 1 ? E_ - 1 : e);
    }
}

// ---------------------------------------------------------------------------
// modReLU elementwise (per-column bias), f32 in -> bf16 out.
// ---------------------------------------------------------------------------
__global__ __launch_bounds__(256) void modrelu_kernel(
    const float* __restrict__ ur, const float* __restrict__ ui,
    const float* __restrict__ mb,
    bf_t* __restrict__ obr, bf_t* __restrict__ obi, size_t n)
{
    size_t i = (size_t)blockIdx.x * 256 + threadIdx.x;
    if (i >= n) return;
    float r = ur[i], im = ui[i];
    float a = sqrtf(r * r + im * im + 1e-10f);
    float s = fmaxf(a + mb[i % HID_], 0.0f) / (a + 1e-10f);
    obr[i] = f2bf(r * s);
    obi[i] = f2bf(im * s);
}

// ---------------------------------------------------------------------------
// LDS-tiled bf16 transpose, 64x64 tiles.  Tile staged into LDS with
// GLOBAL_LOAD_ASYNC_TO_LDS_B128 (+ s_wait_asynccnt) when the toolchain
// exposes the builtins, else plain ds stores.  Both global sides coalesced.
// grid = (C/64, R/64, batches)
// ---------------------------------------------------------------------------
#define TT 64
#define TP 72   // padded row stride (144 B, 16B-aligned rows)

__global__ __launch_bounds__(256) void trbf_kernel(
    const bf_t* __restrict__ src, bf_t* __restrict__ dst, int R, int C)
{
    __shared__ bf_t tile[TT][TP];
    size_t plane = (size_t)R * C;
    const bf_t* s = src + (size_t)blockIdx.z * plane;
    bf_t* d = dst + (size_t)blockIdx.z * plane;
    int c0 = blockIdx.x * TT;
    int r0 = blockIdx.y * TT;
    int t = threadIdx.x;

    // stage 64x64 bf16 tile = 512 x 16B chunks; 8 chunks per row
#ifdef USE_ASYNC_LDS
#pragma unroll
    for (int it = 0; it < 2; ++it) {
        int ch  = t + it * 256;
        int row = ch >> 3;
        int o   = (ch & 7) * 8;
        __builtin_amdgcn_global_load_async_to_lds_b128(
            (AS1 gcc_int4*)(s + (size_t)(r0 + row) * C + c0 + o),
            (AS3 gcc_int4*)&tile[row][o], 0, 0);
    }
    __builtin_amdgcn_s_wait_asynccnt(0);
    __syncthreads();
#else
#pragma unroll
    for (int it = 0; it < 2; ++it) {
        int ch  = t + it * 256;
        int row = ch >> 3;
        int o   = (ch & 7) * 8;
        *(uint4*)&tile[row][o] = *(const uint4*)(s + (size_t)(r0 + row) * C + c0 + o);
    }
    __syncthreads();
#endif

    // write transposed: out element (c, r) = tile[r][c]; consecutive threads
    // walk r within a fixed c  ->  coalesced 2B global stores
#pragma unroll
    for (int it = 0; it < 16; ++it) {
        int f = t + it * 256;
        int c = f >> 6, r = f & 63;
        d[(size_t)(c0 + c) * R + (r0 + r)] = tile[r][c];
    }
}

// ---------------------------------------------------------------------------
// Utility kernels: weight convert+transpose, plain convert, output init.
// ---------------------------------------------------------------------------
__global__ __launch_bounds__(256) void cvt_tr_kernel(
    const float* __restrict__ src, bf_t* __restrict__ dst, int K, int N)
{
    int i = blockIdx.x * 256 + threadIdx.x;
    if (i >= K * N) return;
    int k = i / N, n = i % N;
    dst[(size_t)n * K + k] = f2bf(src[i]);   // [K][N] f32 -> [N][K] bf16
}

__global__ __launch_bounds__(256) void cvt_kernel(
    const float* __restrict__ src, bf_t* __restrict__ dst, size_t n)
{
    size_t i = (size_t)blockIdx.x * 256 + threadIdx.x;
    if (i < n) dst[i] = f2bf(src[i]);
}

__global__ __launch_bounds__(256) void init_kernel(
    const float* __restrict__ a, const float* __restrict__ b,
    float* __restrict__ o, size_t n)
{
    size_t i = (size_t)blockIdx.x * 256 + threadIdx.x;
    if (i >= n) return;
    o[i] = a[i];
    o[n + i] = b[i];
}

// ---------------------------------------------------------------------------
// Host orchestration
// ---------------------------------------------------------------------------
extern "C" void kernel_launch(void* const* d_in, const int* in_sizes, int n_in,
                              void* d_out, int out_size, void* d_ws, size_t ws_size,
                              hipStream_t stream)
{
    (void)in_sizes; (void)n_in; (void)out_size; (void)ws_size;

    char* base = (char*)d_ws;
    size_t off = 0;
    auto alloc = [&](size_t bytes) -> void* {
        void* p = base + off;
        off += (bytes + 255) & ~(size_t)255;
        return p;
    };
    auto F32 = [&](int i) { return (const float*)d_in[i]; };

    const size_t FF  = (size_t)F_ * F_;
    const size_t NF  = (size_t)NTOK * F_;       // 4096*512
    const size_t CN  = (size_t)NCTX * CD_;
    const size_t SCN = (size_t)B_ * NH_ * S_ * S_;
    const size_t NH2 = (size_t)NTOK * HID_;
    const size_t W1sz = (size_t)F_ * HID_;
    const size_t W2sz = (size_t)HID_ * F_;

    // ---- persistent bf16 transposed weights --------------------------------
    bf_t* saT[8];  const int saIdx[8] = {7, 8, 11, 12, 15, 16, 19, 20};
    for (int j = 0; j < 8; ++j) {
        saT[j] = (bf_t*)alloc(FF * 2);
        cvt_tr_kernel<<<(int)((FF + 255) / 256), 256, 0, stream>>>(F32(saIdx[j]), saT[j], F_, F_);
    }
    bf_t* caT[8];  const int caIdx[8] = {23, 24, 27, 28, 31, 32, 35, 36};
    const int caK[8] = {F_, F_, CD_, CD_, CD_, CD_, F_, F_};
    for (int j = 0; j < 8; ++j) {
        size_t cnt = (size_t)caK[j] * F_;
        caT[j] = (bf_t*)alloc(cnt * 2);
        cvt_tr_kernel<<<(int)((cnt + 255) / 256), 256, 0, stream>>>(F32(caIdx[j]), caT[j], caK[j], F_);
    }
    bf_t* w1rT = (bf_t*)alloc((size_t)E_ * W1sz * 2);
    bf_t* w1iT = (bf_t*)alloc((size_t)E_ * W1sz * 2);
    bf_t* w2rT = (bf_t*)alloc((size_t)E_ * W2sz * 2);
    bf_t* w2iT = (bf_t*)alloc((size_t)E_ * W2sz * 2);
    for (int e = 0; e < E_; ++e) {
        cvt_tr_kernel<<<(int)((W1sz + 255) / 256), 256, 0, stream>>>(F32(39) + e * W1sz, w1rT + e * W1sz, F_, HID_);
        cvt_tr_kernel<<<(int)((W1sz + 255) / 256), 256, 0, stream>>>(F32(40) + e * W1sz, w1iT + e * W1sz, F_, HID_);
        cvt_tr_kernel<<<(int)((W2sz + 255) / 256), 256, 0, stream>>>(F32(44) + e * W2sz, w2rT + e * W2sz, HID_, F_);
        cvt_tr_kernel<<<(int)((W2sz + 255) / 256), 256, 0, stream>>>(F32(45) + e * W2sz, w2iT + e * W2sz, HID_, F_);
    }

    // ---- activation scratch ------------------------------------------------
    float* h_r  = (float*)alloc(NF * 4);
    float* h_i  = (float*)alloc(NF * 4);
    bf_t*  h_rb = (bf_t*)alloc(NF * 2);
    bf_t*  h_ib = (bf_t*)alloc(NF * 2);
    bf_t*  q_rb = (bf_t*)alloc(NF * 2);
    bf_t*  q_ib = (bf_t*)alloc(NF * 2);
    bf_t*  k_rb = (bf_t*)alloc(NF * 2);
    bf_t*  k_ib = (bf_t*)alloc(NF * 2);
    bf_t*  v_rb = (bf_t*)alloc(NF * 2);
    bf_t*  v_ib = (bf_t*)alloc(NF * 2);
    bf_t*  vt_rb = (bf_t*)alloc(NF * 2);     // B*F*S == NF
    bf_t*  vt_ib = (bf_t*)alloc(NF * 2);
    float* scores = (float*)alloc(SCN * 4);
    bf_t*  probs  = (bf_t*)alloc(SCN * 2);
    bf_t*  mx_rb = (bf_t*)alloc(NF * 2);
    bf_t*  mx_ib = (bf_t*)alloc(NF * 2);
    float* u_r  = (float*)alloc(NH2 * 4);
    float* u_i  = (float*)alloc(NH2 * 4);
    bf_t*  ub_r = (bf_t*)alloc(NH2 * 2);
    bf_t*  ub_i = (bf_t*)alloc(NH2 * 2);
    bf_t*  ctx_rb = (bf_t*)alloc(CN * 2);
    bf_t*  ctx_ib = (bf_t*)alloc(CN * 2);
    int*   ridx = (int*)alloc(NTOK * 4);

    float* xr = (float*)d_out;
    float* xi = xr + NF;
    init_kernel<<<(int)((NF + 255) / 256), 256, 0, stream>>>(F32(0), F32(1), xr, NF);
    cvt_kernel<<<(int)((CN + 255) / 256), 256, 0, stream>>>(F32(2), ctx_rb, CN);
    cvt_kernel<<<(int)((CN + 255) / 256), 256, 0, stream>>>(F32(3), ctx_ib, CN);

    auto cgemm = [&](const bf_t* Ar, const bf_t* Ai, int lda,
                     const bf_t* Btr, const bf_t* Bti, int ldb,
                     const float* br, const float* bi,
                     float* Cr, float* Ci, bf_t* Crb, bf_t* Cib,
                     const int* ix, int ex, int M, int N, int K, int flags) {
        dim3 g((N + 127) / 128, (M + 63) / 64, 1);
        cgemm_kernel<<<g, 256, 0, stream>>>(Ar, Ai, lda, Btr, Bti, ldb, br, bi,
                                            Cr, Ci, N, Crb, Cib, ix, ex, M, N, K, flags);
    };

    // ====================== Stage 1: self-attention =========================
    cln_kernel<<<NTOK, 256, 0, stream>>>(xr, xi, F32(4), h_r, h_i, h_rb, h_ib);
    cgemm(h_rb, h_ib, F_, saT[0], saT[1], F_, F32(9),  F32(10), nullptr, nullptr, q_rb, q_ib, nullptr, 0, NTOK, F_, F_, CG_BIAS | CG_STBF);
    cgemm(h_rb, h_ib, F_, saT[2], saT[3], F_, F32(13), F32(14), nullptr, nullptr, k_rb, k_ib, nullptr, 0, NTOK, F_, F_, CG_BIAS | CG_STBF);
    cgemm(h_rb, h_ib, F_, saT[4], saT[5], F_, F32(17), F32(18), nullptr, nullptr, v_rb, v_ib, nullptr, 0, NTOK, F_, F_, CG_BIAS | CG_STBF);
    trbf_kernel<<<dim3(F_ / TT, S_ / TT, B_), 256, 0, stream>>>(v_rb, vt_rb, S_, F_);
    trbf_kernel<<<dim3(F_ / TT, S_ / TT, B_), 256, 0, stream>>>(v_ib, vt_ib, S_, F_);
    {
        dim3 g((S_ / 16 + 7) / 8, S_ / 16, B_ * NH_);
        scores_kernel<<<g, 256, 0, stream>>>(q_rb, q_ib, k_rb, k_ib, scores, S_, S_, F_, 0.125f);
    }
    softmax_kernel<<<B_ * NH_ * S_, 256, 0, stream>>>(scores, probs, S_);
    mix_kernel<<<dim3(1, S_ / 16, B_ * NH_), 128, 0, stream>>>(probs, vt_rb, vt_ib, mx_rb, mx_ib, S_, S_);
    cgemm(mx_rb, mx_ib, F_, saT[6], saT[7], F_, F32(21), F32(22), xr, xi, nullptr, nullptr, nullptr, 0, NTOK, F_, F_, CG_BIAS | CG_ACC);

    // ====================== Stage 2: cross-attention ========================
    cln_kernel<<<NTOK, 256, 0, stream>>>(xr, xi, F32(5), h_r, h_i, h_rb, h_ib);
    cgemm(h_rb,  h_ib,  F_,  caT[0], caT[1], F_,  F32(25), F32(26), nullptr, nullptr, q_rb, q_ib, nullptr, 0, NTOK, F_, F_,  CG_BIAS | CG_STBF);
    cgemm(ctx_rb, ctx_ib, CD_, caT[2], caT[3], CD_, F32(29), F32(30), nullptr, nullptr, k_rb, k_ib, nullptr, 0, NCTX, F_, CD_, CG_BIAS | CG_STBF);
    cgemm(ctx_rb, ctx_ib, CD_, caT[4], caT[5], CD_, F32(33), F32(34), nullptr, nullptr, v_rb, v_ib, nullptr, 0, NCTX, F_, CD_, CG_BIAS | CG_STBF);
    trbf_kernel<<<dim3(F_ / TT, L_ / TT, B_), 256, 0, stream>>>(v_rb, vt_rb, L_, F_);
    trbf_kernel<<<dim3(F_ / TT, L_ / TT, B_), 256, 0, stream>>>(v_ib, vt_ib, L_, F_);
    {
        dim3 g((L_ / 16 + 7) / 8, S_ / 16, B_ * NH_);
        scores_kernel<<<g, 256, 0, stream>>>(q_rb, q_ib, k_rb, k_ib, scores, S_, L_, F_, 0.125f);
    }
    softmax_kernel<<<B_ * NH_ * S_, 256, 0, stream>>>(scores, probs, L_);
    mix_kernel<<<dim3(1, S_ / 16, B_ * NH_), 128, 0, stream>>>(probs, vt_rb, vt_ib, mx_rb, mx_ib, S_, L_);
    cgemm(mx_rb, mx_ib, F_, caT[6], caT[7], F_, F32(37), F32(38), xr, xi, nullptr, nullptr, nullptr, 0, NTOK, F_, F_, CG_BIAS | CG_ACC);

    // ====================== Stage 3: phase-routed MoE =======================
    cln_kernel<<<NTOK, 256, 0, stream>>>(xr, xi, F32(6), h_r, h_i, h_rb, h_ib);
    route_kernel<<<NTOK, 256, 0, stream>>>(h_r, h_i, ridx);
    for (int e = 0; e < E_; ++e) {
        cgemm(h_rb, h_ib, F_, w1rT + e * W1sz, w1iT + e * W1sz, F_,
              F32(41) + e * HID_, F32(42) + e * HID_,
              u_r, u_i, nullptr, nullptr, nullptr, 0,
              NTOK, HID_, F_, CG_BIAS | CG_STF32);
        modrelu_kernel<<<(int)((NH2 + 255) / 256), 256, 0, stream>>>(u_r, u_i, F32(43) + e * HID_, ub_r, ub_i, NH2);
        cgemm(ub_r, ub_i, HID_, w2rT + e * W2sz, w2iT + e * W2sz, HID_,
              F32(46) + e * F_, F32(47) + e * F_,
              xr, xi, nullptr, nullptr, ridx, e,
              NTOK, F_, HID_, CG_BIAS | CG_ACC | CG_MASK);
    }
}
